// RingDilatedAttentionMemoryEfficient_64690797412685
// MI455X (gfx1250) — matrix operand order repaired and checked by reading, not compiled, not run on hardware
//
#include <hip/hip_runtime.h>
#include <hip/hip_bf16.h>

// ---------------------------------------------------------------------------
// MHA forward for MI455X (gfx1250): bf16 WMMA GEMMs + flash-attention.
// Reference shapes: B=2, S=2048, E=1024, H=16, Dh=64, causal, scale=1/8.
// ---------------------------------------------------------------------------

typedef __attribute__((ext_vector_type(16))) __bf16 v16bf;
typedef __attribute__((ext_vector_type(8)))  float  v8f;

#define S_LEN   2048
#define EMB     1024
#define NHEAD   16
#define HDIM    64
#define QK_SCALE 0.125f

struct Frag { unsigned int w[8]; };   // 8 VGPRs of packed bf16 pairs (16 elems)

__device__ inline unsigned short f2bf(float f) {
  unsigned int u = __builtin_bit_cast(unsigned int, f);
  unsigned int r = (u + 0x7FFFu + ((u >> 16) & 1u)) >> 16;  // RNE
  return (unsigned short)r;
}

__device__ inline v8f v8zero() {
  v8f z;
#pragma unroll
  for (int i = 0; i < 8; ++i) z[i] = 0.0f;
  return z;
}

__device__ inline v8f wmma_bf16(const Frag& a, const Frag& b, v8f c) {
  return __builtin_amdgcn_wmma_f32_16x16x32_bf16(
      false, __builtin_bit_cast(v16bf, a),
      false, __builtin_bit_cast(v16bf, b),
      (short)0, c, false, false);
}

// ---------------------------------------------------------------------------
// fp32 -> bf16 conversion, 4 elements / thread (all sizes divisible by 1024)
// ---------------------------------------------------------------------------
__global__ __launch_bounds__(256) void cvt_bf16_x4(
    const float* __restrict__ src, unsigned short* __restrict__ dst) {
  size_t i = (size_t)blockIdx.x * blockDim.x + threadIdx.x;
  const float4 f = ((const float4*)src)[i];
  unsigned int lo = (unsigned int)f2bf(f.x) | ((unsigned int)f2bf(f.y) << 16);
  unsigned int hi = (unsigned int)f2bf(f.z) | ((unsigned int)f2bf(f.w) << 16);
  ((uint2*)dst)[i] = make_uint2(lo, hi);
}

// ---------------------------------------------------------------------------
// 128x128 bf16 GEMM, 256 threads = 8 waves (wave32), wave grid 4(M) x 2(N),
// each wave owns 32x64 = 2x4 WMMA accumulators. Software-pipelined:
// global loads for tile k+1 sit in registers while tile k computes;
// registers -> LDS at the iteration boundary (single LDS buffer).
// LDS stride 40 elems: 16B-aligned row starts + conflict-free frag gathers.
// ---------------------------------------------------------------------------
#define LDS_STRIDE 40

struct TileRegs { uint4 a[2]; uint4 b[2]; };

__device__ inline void gemm_fetch(const unsigned short* __restrict__ A,
                                  const unsigned short* __restrict__ B,
                                  int Kdim, int N, int m0, int n0, int k0,
                                  TileRegs& t) {
  const int tid = threadIdx.x;
#pragma unroll
  for (int it = 0; it < 2; ++it) {
    int idx = tid * 2 + it;                 // 0..511
    int r = idx >> 2, c8 = (idx & 3) * 8;   // A tile 128x32
    t.a[it] = *(const uint4*)(A + (size_t)(m0 + r) * Kdim + k0 + c8);
    int rb = idx >> 4, cb8 = (idx & 15) * 8;  // B tile 32x128
    t.b[it] = *(const uint4*)(B + (size_t)(k0 + rb) * N + n0 + cb8);
  }
}

__device__ inline void gemm_stage(const TileRegs& t,
                                  unsigned short* lA, unsigned short* lB) {
  const int tid = threadIdx.x;
#pragma unroll
  for (int it = 0; it < 2; ++it) {
    int idx = tid * 2 + it;
    int r = idx >> 2, c8 = (idx & 3) * 8;
    *(uint4*)(&lA[r * LDS_STRIDE + c8]) = t.a[it];
    int rb = idx >> 4, cb8 = (idx & 15) * 8;
    const unsigned short* e8 = (const unsigned short*)&t.b[it];
#pragma unroll
    for (int j = 0; j < 8; ++j) lB[(cb8 + j) * LDS_STRIDE + rb] = e8[j];
  }
}

__device__ inline void gemm_compute(const unsigned short* lA,
                                    const unsigned short* lB,
                                    int lane, int wm, int wn, v8f acc[2][4]) {
  Frag fa[2], fb[4];
#pragma unroll
  for (int i = 0; i < 2; ++i) {
    int m = wm * 32 + i * 16 + (lane & 15);
#pragma unroll
    for (int v = 0; v < 8; ++v) {            // A-frag 16x32 layout
      int kk = ((v >> 2) << 4) + ((lane >> 4) << 3) + ((v & 3) << 1);
      fa[i].w[v] = *(const unsigned int*)(&lA[m * LDS_STRIDE + kk]);
    }
  }
#pragma unroll
  for (int j = 0; j < 4; ++j) {
    int n = wn * 64 + j * 16 + (lane & 15);
#pragma unroll
    for (int v = 0; v < 8; ++v) {            // B-frag 32x16 layout
      int kk = ((v >> 2) << 4) + ((lane >> 4) << 3) + ((v & 3) << 1);
      fb[j].w[v] = *(const unsigned int*)(&lB[n * LDS_STRIDE + kk]);
    }
  }
#pragma unroll
  for (int i = 0; i < 2; ++i)
#pragma unroll
    for (int j = 0; j < 4; ++j)
      acc[i][j] = wmma_bf16(fa[i], fb[j], acc[i][j]);
}

__device__ inline void gemm_prefetch(const unsigned short* __restrict__ A,
                                     const unsigned short* __restrict__ B,
                                     int Kdim, int N, int m0, int n0, int k0) {
  const int idx = threadIdx.x * 2;
  int r = idx >> 2, c8 = (idx & 3) * 8;
  __builtin_prefetch(A + (size_t)(m0 + r) * Kdim + k0 + c8, 0, 1);
  int rb = idx >> 4, cb8 = (idx & 15) * 8;
  __builtin_prefetch(B + (size_t)(k0 + rb) * N + n0 + cb8, 0, 1);
}

// Kdim must be a multiple of 64 (1024 here).
__device__ inline void gemm_mainloop(const unsigned short* __restrict__ A,
                                     const unsigned short* __restrict__ B,
                                     int Kdim, int N, int m0, int n0,
                                     unsigned short* lA, unsigned short* lB,
                                     v8f acc[2][4]) {
  const int lane = threadIdx.x & 31, wave = threadIdx.x >> 5;
  const int wm = wave >> 1, wn = wave & 1;
  TileRegs ta, tb;
  gemm_fetch(A, B, Kdim, N, m0, n0, 0, ta);
  for (int k0 = 0; k0 < Kdim; k0 += 64) {
    __syncthreads();
    gemm_stage(ta, lA, lB);
    gemm_fetch(A, B, Kdim, N, m0, n0, k0 + 32, tb);
    if (k0 + 64 < Kdim) gemm_prefetch(A, B, Kdim, N, m0, n0, k0 + 64);
    __syncthreads();
    gemm_compute(lA, lB, lane, wm, wn, acc);

    __syncthreads();
    gemm_stage(tb, lA, lB);
    if (k0 + 64 < Kdim) gemm_fetch(A, B, Kdim, N, m0, n0, k0 + 64, ta);
    if (k0 + 96 < Kdim) gemm_prefetch(A, B, Kdim, N, m0, n0, k0 + 96);
    __syncthreads();
    gemm_compute(lA, lB, lane, wm, wn, acc);
  }
}

// ---------------------------------------------------------------------------
// Kernel 1: fused QKV projection. C = Xb @ Wqkv + b, scattered into
// Q/K/V [B,H,S,64] bf16; Q pre-scaled by 1/sqrt(Dh).
// The 128-col tile lies entirely inside one of Q/K/V and one batch:
// destination pointer / scale / batch are block-uniform scalars.
// grid (3072/128, 4096/128)
// ---------------------------------------------------------------------------
__global__ __launch_bounds__(256) void gemm_qkv_kernel(
    const unsigned short* __restrict__ X, const unsigned short* __restrict__ W,
    const float* __restrict__ bias,
    unsigned short* __restrict__ Qd, unsigned short* __restrict__ Kd,
    unsigned short* __restrict__ Vd) {
  __shared__ unsigned short lA[128 * LDS_STRIDE];
  __shared__ unsigned short lB[128 * LDS_STRIDE];
  const int lane = threadIdx.x & 31, wave = threadIdx.x >> 5;
  const int wm = wave >> 1, wn = wave & 1;
  const int m0 = blockIdx.y * 128, n0 = blockIdx.x * 128;

  v8f acc[2][4];
#pragma unroll
  for (int i = 0; i < 2; ++i)
#pragma unroll
    for (int j = 0; j < 4; ++j) acc[i][j] = v8zero();

  gemm_mainloop(X, W, EMB, 3 * EMB, m0, n0, lA, lB, acc);

  const int sel = n0 >> 10;                         // block-uniform: 0/1/2
  unsigned short* const Dst = (sel == 0) ? Qd : ((sel == 1) ? Kd : Vd);
  const float oscale = (sel == 0) ? QK_SCALE : 1.0f;
  const int b  = m0 >> 11;                          // block-uniform batch
  const int s0 = m0 & 2047;

#pragma unroll
  for (int i = 0; i < 2; ++i)
#pragma unroll
    for (int j = 0; j < 4; ++j) {
      const int n = n0 + wn * 64 + j * 16 + (lane & 15);
      const float bj = bias[n];
      const int nn = n & 1023;
      const int h = nn >> 6, d = nn & 63;
      const size_t cbase = (size_t)(b * NHEAD + h) * (S_LEN * HDIM) + d;
      const int srow = s0 + wm * 32 + i * 16 + ((lane >> 4) << 3);
#pragma unroll
      for (int e = 0; e < 8; ++e) {
        float val = (acc[i][j][e] + bj) * oscale;
        Dst[cbase + (size_t)(srow + e) * HDIM] = f2bf(val);
      }
    }
}

// ---------------------------------------------------------------------------
// Kernel 2: causal flash attention. One workgroup per (b*h, 128 q rows);
// 8 waves x 16 q rows each. K/V tiles (32 keys) software-pipelined through
// registers into LDS; scores stay in accumulators; online softmax via
// 16-lane xor-shuffles; P relaid through wave-private LDS into A-frag format.
// grid (2048/128, B*H)
// ---------------------------------------------------------------------------
__global__ __launch_bounds__(256) void attn_kernel(
    const unsigned short* __restrict__ Q, const unsigned short* __restrict__ K,
    const unsigned short* __restrict__ V, unsigned short* __restrict__ AO) {
  __shared__ unsigned short lK[32 * 72];       // [key][dh], stride 72
  __shared__ unsigned short lVt[64 * 34];      // [dh][key], stride 34
  __shared__ unsigned short lP[8][16 * 34];    // per-wave P scratch, stride 34
  const int tid = threadIdx.x, lane = tid & 31, wave = tid >> 5;
  const int bh = blockIdx.y;
  const int qwg = blockIdx.x * 128;
  const int qb = qwg + wave * 16;
  const unsigned short* Qb = Q + (size_t)bh * S_LEN * HDIM;
  const unsigned short* Kb = K + (size_t)bh * S_LEN * HDIM;
  const unsigned short* Vb = V + (size_t)bh * S_LEN * HDIM;

  // Q fragments (Dh=64 -> two 16x32 A-frags), loaded once from global.
  Frag fq[2];
#pragma unroll
  for (int d = 0; d < 2; ++d) {
    int m = qb + (lane & 15);
#pragma unroll
    for (int v = 0; v < 8; ++v) {
      int kk = ((v >> 2) << 4) + ((lane >> 4) << 3) + ((v & 3) << 1) + d * 32;
      fq[d].w[v] = *(const unsigned int*)(Qb + (size_t)m * HDIM + kk);
    }
  }

  v8f accO[4];
#pragma unroll
  for (int j = 0; j < 4; ++j) accO[j] = v8zero();
  float mi[8], li[8];
#pragma unroll
  for (int e = 0; e < 8; ++e) { mi[e] = -1.0e30f; li[e] = 0.0f; }

  const int nkt_wave = (qb + 47) >> 5;              // causal tile count (wave)
  const int nkt_wg   = (qwg + 112 + 47) >> 5;       // max over workgroup

  // staging coordinates (cooperative: 256 threads x 16B per array)
  const int sr = tid >> 3, sc8 = (tid & 7) * 8;

  uint4 kc = *(const uint4*)(Kb + (size_t)sr * HDIM + sc8);
  uint4 vc = *(const uint4*)(Vb + (size_t)sr * HDIM + sc8);
  uint4 kn, vn;

  for (int kt = 0; kt < nkt_wg; ++kt) {
    const int kt0 = kt * 32;
    __syncthreads();
    {  // stage K (row-major) and V (transposed) tiles from registers
      *(uint4*)(&lK[sr * 72 + sc8]) = kc;
      const unsigned short* e8 = (const unsigned short*)&vc;
#pragma unroll
      for (int j = 0; j < 8; ++j) lVt[(sc8 + j) * 34 + sr] = e8[j];
    }
    const bool more = (kt + 1 < nkt_wg);
    if (more) {
      kn = *(const uint4*)(Kb + (size_t)(kt0 + 32 + sr) * HDIM + sc8);
      vn = *(const uint4*)(Vb + (size_t)(kt0 + 32 + sr) * HDIM + sc8);
    }
    if (kt + 2 < nkt_wg) {
      __builtin_prefetch(Kb + (size_t)(kt0 + 64 + sr) * HDIM + sc8, 0, 1);
      __builtin_prefetch(Vb + (size_t)(kt0 + 64 + sr) * HDIM + sc8, 0, 1);
    }
    __syncthreads();

    if (kt < nkt_wave) {   // wave-uniform: EXEC stays all-ones around WMMA
      // scores: S(16x32) = Q(16x64) @ K^T(64x32), two 16-key column chunks
      v8f sc[2];
#pragma unroll
      for (int c = 0; c < 2; ++c) {
        Frag fk0, fk1;
        int key = c * 16 + (lane & 15);
#pragma unroll
        for (int v = 0; v < 8; ++v) {
          int kk = ((v >> 2) << 4) + ((lane >> 4) << 3) + ((v & 3) << 1);
          fk0.w[v] = *(const unsigned int*)(&lK[key * 72 + kk]);
          fk1.w[v] = *(const unsigned int*)(&lK[key * 72 + 32 + kk]);
        }
        v8f z = v8zero();
        z = wmma_bf16(fq[0], fk0, z);
        z = wmma_bf16(fq[1], fk1, z);
        sc[c] = z;
      }

      if (kt0 + 31 > qb) {  // diagonal tile: causal mask
#pragma unroll
        for (int c = 0; c < 2; ++c)
#pragma unroll
          for (int e = 0; e < 8; ++e) {
            int key = kt0 + c * 16 + (lane & 15);
            int qrow = qb + ((lane >> 4) << 3) + e;
            if (key > qrow) sc[c][e] = -3.0e38f;
          }
      }

      // online softmax (row = VGPR index e within each 16-lane half-group)
#pragma unroll
      for (int e = 0; e < 8; ++e) {
        float rm = fmaxf(sc[0][e], sc[1][e]);
#pragma unroll
        for (int off = 1; off < 16; off <<= 1)
          rm = fmaxf(rm, __shfl_xor(rm, off, 32));
        float mnew = fmaxf(mi[e], rm);
        float corr = __expf(mi[e] - mnew);
        mi[e] = mnew;
        float p0 = __expf(sc[0][e] - mnew);
        float p1 = __expf(sc[1][e] - mnew);
        sc[0][e] = p0; sc[1][e] = p1;
        float rs = p0 + p1;
#pragma unroll
        for (int off = 1; off < 16; off <<= 1)
          rs += __shfl_xor(rs, off, 32);
        li[e] = li[e] * corr + rs;
#pragma unroll
        for (int j = 0; j < 4; ++j) accO[j][e] *= corr;
      }

      // P (C-layout) -> LDS bf16 -> reload in A-frag layout
      unsigned short* myP = &lP[wave][0];
#pragma unroll
      for (int c = 0; c < 2; ++c)
#pragma unroll
        for (int e = 0; e < 8; ++e)
          myP[(((lane >> 4) << 3) + e) * 34 + c * 16 + (lane & 15)] =
              f2bf(sc[c][e]);
      asm volatile("s_wait_dscnt 0" ::: "memory");
      Frag fp;
      {
        int m = lane & 15;
#pragma unroll
        for (int v = 0; v < 8; ++v) {
          int kk = ((v >> 2) << 4) + ((lane >> 4) << 3) + ((v & 3) << 1);
          fp.w[v] = *(const unsigned int*)(myP + m * 34 + kk);
        }
      }
      // O += P(16x32) @ V(32x64)
#pragma unroll
      for (int j = 0; j < 4; ++j) {
        Frag fv;
        int n = j * 16 + (lane & 15);
#pragma unroll
        for (int v = 0; v < 8; ++v) {
          int kk = ((v >> 2) << 4) + ((lane >> 4) << 3) + ((v & 3) << 1);
          fv.w[v] = *(const unsigned int*)(&lVt[n * 34 + kk]);
        }
        accO[j] = wmma_bf16(fp, fv, accO[j]);
      }
    }
    if (more) { kc = kn; vc = vn; }
  }

  // normalize + write [B,S,E] bf16 (head h occupies cols h*64..h*64+63)
  const int b = bh >> 4, h = bh & 15;
#pragma unroll
  for (int e = 0; e < 8; ++e) {
    float inv = 1.0f / li[e];
    int s = qb + ((lane >> 4) << 3) + e;
#pragma unroll
    for (int j = 0; j < 4; ++j) {
      int col = h * HDIM + j * 16 + (lane & 15);
      AO[((size_t)(b * S_LEN + s)) * EMB + col] = f2bf(accO[j][e] * inv);
    }
  }
}

// ---------------------------------------------------------------------------
// Kernel 3: output projection, f32 epilogue straight to d_out.
// grid (1024/128, 4096/128)
// ---------------------------------------------------------------------------
__global__ __launch_bounds__(256) void gemm_out_kernel(
    const unsigned short* __restrict__ A, const unsigned short* __restrict__ W,
    const float* __restrict__ bias, float* __restrict__ out) {
  __shared__ unsigned short lA[128 * LDS_STRIDE];
  __shared__ unsigned short lB[128 * LDS_STRIDE];
  const int lane = threadIdx.x & 31, wave = threadIdx.x >> 5;
  const int wm = wave >> 1, wn = wave & 1;
  const int m0 = blockIdx.y * 128, n0 = blockIdx.x * 128;

  v8f acc[2][4];
#pragma unroll
  for (int i = 0; i < 2; ++i)
#pragma unroll
    for (int j = 0; j < 4; ++j) acc[i][j] = v8zero();

  gemm_mainloop(A, W, EMB, EMB, m0, n0, lA, lB, acc);

#pragma unroll
  for (int i = 0; i < 2; ++i)
#pragma unroll
    for (int j = 0; j < 4; ++j) {
      const int n = n0 + wn * 64 + j * 16 + (lane & 15);
      const float bj = bias[n];
      const int m0w = m0 + wm * 32 + i * 16 + ((lane >> 4) << 3);
#pragma unroll
      for (int e = 0; e < 8; ++e)
        out[(size_t)(m0w + e) * EMB + n] = acc[i][j][e] + bj;
    }
}

// ---------------------------------------------------------------------------
extern "C" void kernel_launch(void* const* d_in, const int* in_sizes, int n_in,
                              void* d_out, int out_size, void* d_ws,
                              size_t ws_size, hipStream_t stream) {
  (void)in_sizes; (void)n_in; (void)out_size; (void)ws_size;
  const float* x     = (const float*)d_in[0];  // [2,2048,1024]
  const float* w_qkv = (const float*)d_in[1];  // [1024,3072]
  const float* b_qkv = (const float*)d_in[2];  // [3072]
  const float* w_out = (const float*)d_in[3];  // [1024,1024]
  const float* b_out = (const float*)d_in[4];  // [1024]
  float* out = (float*)d_out;                  // [2,2048,1024]

  char* ws = (char*)d_ws;  // 48 MiB layout
  unsigned short* xb    = (unsigned short*)(ws);
  unsigned short* wqkvb = (unsigned short*)(ws + (size_t)(8u)  * 1048576u);
  unsigned short* woutb = (unsigned short*)(ws + (size_t)(14u) * 1048576u);
  unsigned short* q     = (unsigned short*)(ws + (size_t)(16u) * 1048576u);
  unsigned short* k     = (unsigned short*)(ws + (size_t)(24u) * 1048576u);
  unsigned short* v     = (unsigned short*)(ws + (size_t)(32u) * 1048576u);
  unsigned short* ao    = (unsigned short*)(ws + (size_t)(40u) * 1048576u);

  cvt_bf16_x4<<<4096, 256, 0, stream>>>(x, xb);        // 4,194,304 elems
  cvt_bf16_x4<<<3072, 256, 0, stream>>>(w_qkv, wqkvb); // 3,145,728 elems
  cvt_bf16_x4<<<1024, 256, 0, stream>>>(w_out, woutb); // 1,048,576 elems

  gemm_qkv_kernel<<<dim3(24, 32), 256, 0, stream>>>(xb, wqkvb, b_qkv, q, k, v);
  attn_kernel<<<dim3(16, 32), 256, 0, stream>>>(q, k, v, ao);
  gemm_out_kernel<<<dim3(8, 32), 256, 0, stream>>>(ao, woutb, b_out, out);
}
